// disciminativeAno_35158602285285
// MI455X (gfx1250) — matrix operand order repaired and checked by reading, not compiled, not run on hardware
//
#include <hip/hip_runtime.h>
#include <hip/hip_bf16.h>

// ---------------------------------------------------------------------------
// VAE anomaly pipeline for MI455X (gfx1250, wave32, WMMA).
// All conv / convT / FC layers run through one implicit-GEMM kernel using
// v_wmma_f32_16x16x32_f16 (f16 inputs, f32 accumulate). Transposed convs are
// expressed as dilated-input convolutions handled inside the im2col gather
// (no materialized zero-upsampled buffers). BN folded to scale/shift,
// epilogues fused (templated). Double-buffered LDS: one barrier per K-step,
// global->LDS staging of step k+1 overlaps WMMA of step k. Incremental tap
// state avoids per-step integer division.
//
// Input pointer order assumption: setup_inputs() dict order, each BN tuple
// flattened to 4 consecutive arrays (g, beta, mean, var) -> 46 inputs total.
// Workspace use: ~379 MB of d_ws (f16 activations + packed f16 weights).
// ---------------------------------------------------------------------------

typedef __attribute__((ext_vector_type(16))) _Float16 v16h;
typedef __attribute__((ext_vector_type(8)))  float    v8f;
typedef __attribute__((ext_vector_type(4)))  unsigned u32x4;

union HFrag { u32x4 q[2]; v16h h; };   // trivial types only -> legal union

// --------------------------------------------------------------------------
// Weight pack: src f32 [Kreal][Cout] (i.e. [KH][KW][Cin][Cout] flattened)
// -> dst f16 [Coutp][Kpad], zero padded.  k = (ky*KW+kx)*Cin + ci.
// --------------------------------------------------------------------------
__global__ __launch_bounds__(256) void pack_w(const float* __restrict__ src,
                                              _Float16* __restrict__ dst,
                                              int Kreal, int Kpad, int Cout,
                                              int Coutp) {
  int i = blockIdx.x * 256 + threadIdx.x;
  int total = Coutp * Kpad;
  if (i >= total) return;
  int co = i / Kpad, k = i - co * Kpad;
  float v = (k < Kreal && co < Cout) ? src[(long)k * Cout + co] : 0.0f;
  dst[i] = (_Float16)v;
}

// BN fold: scale = g*rsqrt(v+eps), shift = beta - mean*scale
__global__ __launch_bounds__(128) void prep_bn(const float* __restrict__ g,
                                               const float* __restrict__ be,
                                               const float* __restrict__ mm,
                                               const float* __restrict__ vv,
                                               float* __restrict__ scale,
                                               float* __restrict__ shift, int C) {
  int c = threadIdx.x;
  if (c < C) {
    float s = g[c] * rsqrtf(vv[c] + 1e-3f);
    scale[c] = s;
    shift[c] = be[c] - mm[c] * s;
  }
}

__global__ __launch_bounds__(64) void pad_bias(const float* __restrict__ src,
                                               float* __restrict__ dst,
                                               int n, int np) {
  int i = threadIdx.x;
  if (i < np) dst[i] = (i < n) ? src[i] : 0.0f;
}

__global__ __launch_bounds__(256) void f32_to_f16(const float* __restrict__ s,
                                                  _Float16* __restrict__ d,
                                                  long n) {
  long i = (long)blockIdx.x * 256 + threadIdx.x;
  if (i < n) d[i] = (_Float16)s[i];
}

// z = z_mean + exp(0.5*z_log_var) * eps
__global__ __launch_bounds__(256) void zsample(const _Float16* __restrict__ zm,
                                               const _Float16* __restrict__ zlv,
                                               const float* __restrict__ eps,
                                               _Float16* __restrict__ z, long n) {
  long i = (long)blockIdx.x * 256 + threadIdx.x;
  if (i < n) {
    float v = (float)zm[i] + __expf(0.5f * (float)zlv[i]) * eps[i];
    z[i] = (_Float16)v;
  }
}

// --------------------------------------------------------------------------
// Implicit-GEMM conv via WMMA, double-buffered LDS software pipeline.
//   out[n,oy,ox,co] = epilogue( sum_{ky,kx,ci} in[n,iy,ix,ci]*W[k,co] + b[co] )
//   iy_num = oy*stride + ky - pad ; valid iff iy_num % dil == 0 and in range.
// Block: 128 threads (4 wave32). Block tile: M=64, N=16*NFRAG, K step 32.
// GENERIC=true: per-element gather (enc1, Cin=3, K padded 75->96).
// GENERIC=false: Cin%32==0, Kreal==Kpad; 16-half vector gather, incremental
//                (ci0,kx,ky) tap state (no divisions in the loop).
// MODE 0: relu -> BN(scale,shift) ; 1: relu ; 2: sigmoid.
// --------------------------------------------------------------------------
template <int NFRAG, bool GENERIC, int MODE>
__global__ __launch_bounds__(128) void conv_gemm(
    const _Float16* __restrict__ in, const _Float16* __restrict__ wt,
    _Float16* __restrict__ out, const float* __restrict__ bias,
    const float* __restrict__ scale, const float* __restrict__ shift,
    int IH, int IW, int Cin, int OH, int OW, int KH, int KW, int stride,
    int pad, int dil, int Kreal, int Kpad, int Couts) {
  __shared__ _Float16 As[2][64 * 40];          // 64 M-rows x 32 K (stride 40)
  __shared__ _Float16 Bs[2][NFRAG * 16 * 40];  // N cols x 32 K (stride 40)

  const int t = threadIdx.x;
  const int lane = t & 31;
  const int wv = t >> 5;
  const long m0 = (long)blockIdx.x * 64;
  const int n0 = blockIdx.y * (NFRAG * 16);

  // Row this thread stages into As (row = t>>1, half-chunk = (t&1)*16)
  const int arow = t >> 1;
  const int acol0 = (t & 1) * 16;
  const long r = m0 + arow;
  const int ospat = OH * OW;
  const int n_img = (int)(r / ospat);
  const int remsp = (int)(r - (long)n_img * ospat);
  const int oy = remsp / OW, ox = remsp - (remsp / OW) * OW;
  const int by = oy * stride - pad, bx = ox * stride - pad;
  const long in_img = (long)n_img * IH * IW * Cin;  // image base (elements)

  // Incremental tap state for the fast path: k0 = kc + acol0, acol0 < 32 <= Cin.
  int ci0 = acol0, kx = 0, ky = 0;

  v8f acc[NFRAG];
#pragma unroll
  for (int f = 0; f < NFRAG; ++f)
#pragma unroll
    for (int v = 0; v < 8; ++v) acc[f][v] = 0.0f;

  // Per-thread fixed LDS addresses
  _Float16* asdst0 = &As[0][arow * 40 + acol0];
  _Float16* asdst1 = &As[1][arow * 40 + acol0];

  const int steps = Kpad >> 5;  // Kpad / 32

  // ---- staging helpers ----
  auto stageA = [&](int kc, _Float16* dst) {
    if (GENERIC) {
      for (int j = 0; j < 16; ++j) {
        int k = kc + acol0 + j;
        _Float16 v = (_Float16)0.0f;
        if (k < Kreal) {
          int kk = k / Cin;
          int ci = k - kk * Cin;
          int gkx = kk % KW, gky = kk / KW;
          int iyn = by + gky, ixn = bx + gkx;
          if (((iyn % dil) == 0) && ((ixn % dil) == 0)) {
            int iy = iyn / dil, ix = ixn / dil;
            if (iy >= 0 && iy < IH && ix >= 0 && ix < IW)
              v = in[in_img + ((long)iy * IW + ix) * Cin + ci];
          }
        }
        dst[j] = v;
      }
    } else {
      // tap (ky,kx) constant across this thread's 16 contiguous k's
      int iyn = by + ky, ixn = bx + kx;
      bool valid = ((iyn % dil) == 0) && ((ixn % dil) == 0);
      int iy = iyn / dil, ix = ixn / dil;
      valid = valid && iy >= 0 && iy < IH && ix >= 0 && ix < IW;
      u32x4* d4 = (u32x4*)dst;
      if (valid) {
        const u32x4* s4 =
            (const u32x4*)(in + in_img + ((long)iy * IW + ix) * Cin + ci0);
        d4[0] = s4[0];
        d4[1] = s4[1];
      } else {
        u32x4 zz = 0;
        d4[0] = zz;
        d4[1] = zz;
      }
      // advance tap state to kc+32 (Cin % 32 == 0 -> wraps at most once)
      ci0 += 32;
      if (ci0 >= Cin) {
        ci0 -= Cin;
        ++kx;
        if (kx == KW) { kx = 0; ++ky; }
      }
    }
  };
  auto stageB = [&](int kc, _Float16* bsbase) {
#pragma unroll
    for (int c = t; c < NFRAG * 16 * 2; c += 128) {
      int brow = c >> 1;
      int bc0 = (c & 1) * 16;
      const u32x4* s = (const u32x4*)(wt + (long)(n0 + brow) * Kpad + kc + bc0);
      u32x4* d = (u32x4*)(bsbase + brow * 40 + bc0);
      d[0] = s[0];
      d[1] = s[1];
    }
  };

  // ---- pipeline prologue ----
  stageA(0, asdst0);
  stageB(0, Bs[0]);
  __syncthreads();

  const int afrow = wv * 16 + (lane & 15);
  const int abase = (lane < 16) ? 0 : 8;   // A: K chunks [abase,+8) & [abase+16,+8)
  const int kb = (lane < 16) ? 0 : 8;      // B: K chunks [kb,+8) & [kb+16,+8)

  for (int s = 0; s < steps; ++s) {
    const int cur = s & 1;
    if (s + 1 < steps) {
      __builtin_prefetch(wt + (long)n0 * Kpad + (s + 2) * 32, 0, 1);
      stageA((s + 1) * 32, cur ? asdst0 : asdst1);
      stageB((s + 1) * 32, Bs[cur ^ 1]);
    }
    // ---- fragments per documented 16-bit WMMA layouts ----
    HFrag af;
    af.q[0] = *(const u32x4*)(&As[cur][afrow * 40 + abase]);
    af.q[1] = *(const u32x4*)(&As[cur][afrow * 40 + abase + 16]);
#pragma unroll
    for (int f = 0; f < NFRAG; ++f) {
      HFrag bf;
      int col = f * 16 + (lane & 15);
      bf.q[0] = *(const u32x4*)(&Bs[cur][col * 40 + kb]);
      bf.q[1] = *(const u32x4*)(&Bs[cur][col * 40 + kb + 16]);
      acc[f] = __builtin_amdgcn_wmma_f32_16x16x32_f16(
          false, af.h, false, bf.h, (short)0, acc[f], false, false);
    }
    __syncthreads();
  }

  // ---- fused epilogue + f16 store ----
  const long mrow = m0 + wv * 16 + ((lane < 16) ? 0 : 8);
#pragma unroll
  for (int f = 0; f < NFRAG; ++f) {
    int col = n0 + f * 16 + (lane & 15);
    float b = bias[col];
    float sc = (MODE == 0) ? scale[col] : 0.0f;
    float sh = (MODE == 0) ? shift[col] : 0.0f;
#pragma unroll
    for (int v = 0; v < 8; ++v) {
      float c = acc[f][v] + b;
      float res;
      if (MODE == 0) {
        res = fmaxf(c, 0.0f) * sc + sh;
      } else if (MODE == 1) {
        res = fmaxf(c, 0.0f);
      } else {
        res = 1.0f / (1.0f + __expf(-c));
      }
      out[(mrow + v) * (long)Couts + col] = (_Float16)res;
    }
  }
}

// --------------------------------------------------------------------------
// Fused per-image error + Otsu threshold + mask. One block per batch image.
// error = mean_c (x - recon)^2 ; bins = searchsorted(arange(0,.91,.01), e, 'right')
// --------------------------------------------------------------------------
__global__ __launch_bounds__(256) void otsu_mask(const float* __restrict__ x,
                                                 const _Float16* __restrict__ recon,
                                                 int* __restrict__ out) {
  __shared__ float err[1024];
  __shared__ float sums[92];
  __shared__ unsigned counts[92];
  __shared__ float thr_s;
  const int b = blockIdx.x, t = threadIdx.x;
  if (t < 92) { sums[t] = 0.0f; counts[t] = 0u; }
  __syncthreads();
  for (int p = t; p < 1024; p += 256) {
    long px = (long)b * 1024 + p;
    float e = 0.0f;
#pragma unroll
    for (int c = 0; c < 3; ++c) {
      float d = x[px * 3 + c] - (float)recon[px * 16 + c];
      e += d * d;
    }
    e *= (1.0f / 3.0f);
    err[p] = e;
    int bin = (int)(e * 100.0f) + 1;  // #{k : k*0.01 <= e}, e in [0,1]
    if (bin > 91) bin = 91;
    atomicAdd(&counts[bin], 1u);
    atomicAdd(&sums[bin], e);
  }
  __syncthreads();
  if (t == 0) {
    float tot = 0.0f;
    for (int k = 0; k < 92; ++k) tot += sums[k];
    float best = -1.0f, cc = 0.0f, ss = 0.0f;
    int bi = 0;
    for (int k = 0; k < 91; ++k) {
      cc += (float)counts[k];
      ss += sums[k];
      float sig = 0.0f;
      if (cc > 0.0f && cc < 1024.0f) {
        float p0 = cc * (1.0f / 1024.0f);
        float m0 = ss / cc;
        float m1 = (tot - ss) / (1024.0f - cc);
        float d = m0 - m1;
        sig = p0 * (1.0f - p0) * d * d;
      }
      if (sig > best) { best = sig; bi = k; }  // first argmax
    }
    thr_s = 0.01f * (float)bi;
  }
  __syncthreads();
  float thr = thr_s;
  for (int p = t; p < 1024; p += 256)
    out[(long)b * 1024 + p] = (err[p] < thr) ? 1 : 0;
}

// --------------------------------------------------------------------------
extern "C" void kernel_launch(void* const* d_in, const int* in_sizes, int n_in,
                              void* d_out, int out_size, void* d_ws,
                              size_t ws_size, hipStream_t stream) {
  (void)in_sizes; (void)n_in; (void)out_size; (void)ws_size;
  const int B = 2048;

  // ---- inputs (dict order, BN tuples flattened g,beta,mean,var) ----
  const float* x     = (const float*)d_in[0];
  const float* eps_z = (const float*)d_in[1];
  const float* e_w1  = (const float*)d_in[2];
  const float* e_b1  = (const float*)d_in[3];
  const float* bn1g = (const float*)d_in[4],  *bn1b = (const float*)d_in[5];
  const float* bn1m = (const float*)d_in[6],  *bn1v = (const float*)d_in[7];
  const float* e_w2  = (const float*)d_in[8];
  const float* e_b2  = (const float*)d_in[9];
  const float* bn2g = (const float*)d_in[10], *bn2b = (const float*)d_in[11];
  const float* bn2m = (const float*)d_in[12], *bn2v = (const float*)d_in[13];
  const float* e_w3  = (const float*)d_in[14];
  const float* e_b3  = (const float*)d_in[15];
  const float* bn3g = (const float*)d_in[16], *bn3b = (const float*)d_in[17];
  const float* bn3m = (const float*)d_in[18], *bn3v = (const float*)d_in[19];
  const float* e_wm  = (const float*)d_in[20];
  const float* e_bm  = (const float*)d_in[21];
  const float* e_wv  = (const float*)d_in[22];
  const float* e_bv  = (const float*)d_in[23];
  const float* d_wd  = (const float*)d_in[24];
  const float* d_bd  = (const float*)d_in[25];
  const float* d_w1  = (const float*)d_in[26];
  const float* d_b1  = (const float*)d_in[27];
  const float* dn1g = (const float*)d_in[28], *dn1b = (const float*)d_in[29];
  const float* dn1m = (const float*)d_in[30], *dn1v = (const float*)d_in[31];
  const float* d_w2  = (const float*)d_in[32];
  const float* d_b2  = (const float*)d_in[33];
  const float* dn2g = (const float*)d_in[34], *dn2b = (const float*)d_in[35];
  const float* dn2m = (const float*)d_in[36], *dn2v = (const float*)d_in[37];
  const float* d_w3  = (const float*)d_in[38];
  const float* d_b3  = (const float*)d_in[39];
  const float* dn3g = (const float*)d_in[40], *dn3b = (const float*)d_in[41];
  const float* dn3m = (const float*)d_in[42], *dn3v = (const float*)d_in[43];
  const float* d_w4  = (const float*)d_in[44];
  const float* d_b4  = (const float*)d_in[45];
  int* mask_out = (int*)d_out;

  // ---- workspace carve-out ----
  char* ws = (char*)d_ws;
  size_t off = 0;
  auto alloc = [&](size_t bytes) -> char* {
    char* p = ws + off;
    off = (off + bytes + 255) & ~(size_t)255;
    return p;
  };
  auto allocH = [&](size_t n) -> _Float16* { return (_Float16*)alloc(n * 2); };
  auto allocF = [&](size_t n) -> float*    { return (float*)alloc(n * 4); };

  // packed f16 weights [Coutp][Kpad]
  _Float16* wt1 = allocH(32 * 96);
  _Float16* wt2 = allocH(64 * 800);
  _Float16* wt3 = allocH((size_t)128 * 1600);
  _Float16* wtm = allocH((size_t)128 * 2048);
  _Float16* wtv = allocH((size_t)128 * 2048);
  _Float16* wtd = allocH((size_t)8192 * 128);
  _Float16* wtD1 = allocH(128 * 128);
  _Float16* wtD2 = allocH(64 * 1152);
  _Float16* wtD3 = allocH(32 * 576);
  _Float16* wtD4 = allocH(16 * 288);
  // BN scale/shift + padded final bias
  float* sc1 = allocF(32);  float* sh1 = allocF(32);
  float* sc2 = allocF(64);  float* sh2 = allocF(64);
  float* sc3 = allocF(128); float* sh3 = allocF(128);
  float* scD1 = allocF(128); float* shD1 = allocF(128);
  float* scD2 = allocF(64);  float* shD2 = allocF(64);
  float* scD3 = allocF(32);  float* shD3 = allocF(32);
  float* b4p = allocF(16);
  // f16 activations (a1 reused as d1; d2 buffer lifetime-disjoint from recon)
  _Float16* xh  = allocH((size_t)B * 32 * 32 * 3);
  _Float16* a1  = allocH((size_t)B * 16 * 16 * 32);   // also d1 [B,8,8,128]
  _Float16* a2  = allocH((size_t)B * 8 * 8 * 64);
  _Float16* a3  = allocH((size_t)B * 2048);
  _Float16* zm  = allocH((size_t)B * 128);
  _Float16* zlv = allocH((size_t)B * 128);
  _Float16* z   = allocH((size_t)B * 128);
  _Float16* d0  = allocH((size_t)B * 8192);
  _Float16* d2  = allocH((size_t)B * 16 * 16 * 64);
  _Float16* d3  = allocH((size_t)B * 32 * 32 * 32);
  _Float16* recon = allocH((size_t)B * 32 * 32 * 16); // Cout padded 3->16
  _Float16* d1 = a1;

  auto cdiv = [](long a, long b) { return (int)((a + b - 1) / b); };

  // ---- weight / BN prep (cheap, deterministic every call) ----
  pack_w<<<cdiv(32 * 96, 256), 256, 0, stream>>>(e_w1, wt1, 75, 96, 32, 32);
  pack_w<<<cdiv(64 * 800, 256), 256, 0, stream>>>(e_w2, wt2, 800, 800, 64, 64);
  pack_w<<<cdiv(128 * 1600, 256), 256, 0, stream>>>(e_w3, wt3, 1600, 1600, 128, 128);
  pack_w<<<cdiv(128 * 2048, 256), 256, 0, stream>>>(e_wm, wtm, 2048, 2048, 128, 128);
  pack_w<<<cdiv(128 * 2048, 256), 256, 0, stream>>>(e_wv, wtv, 2048, 2048, 128, 128);
  pack_w<<<cdiv(8192 * 128, 256), 256, 0, stream>>>(d_wd, wtd, 128, 128, 8192, 8192);
  pack_w<<<cdiv(128 * 128, 256), 256, 0, stream>>>(d_w1, wtD1, 128, 128, 128, 128);
  pack_w<<<cdiv(64 * 1152, 256), 256, 0, stream>>>(d_w2, wtD2, 1152, 1152, 64, 64);
  pack_w<<<cdiv(32 * 576, 256), 256, 0, stream>>>(d_w3, wtD3, 576, 576, 32, 32);
  pack_w<<<cdiv(16 * 288, 256), 256, 0, stream>>>(d_w4, wtD4, 288, 288, 3, 16);
  prep_bn<<<1, 128, 0, stream>>>(bn1g, bn1b, bn1m, bn1v, sc1, sh1, 32);
  prep_bn<<<1, 128, 0, stream>>>(bn2g, bn2b, bn2m, bn2v, sc2, sh2, 64);
  prep_bn<<<1, 128, 0, stream>>>(bn3g, bn3b, bn3m, bn3v, sc3, sh3, 128);
  prep_bn<<<1, 128, 0, stream>>>(dn1g, dn1b, dn1m, dn1v, scD1, shD1, 128);
  prep_bn<<<1, 128, 0, stream>>>(dn2g, dn2b, dn2m, dn2v, scD2, shD2, 64);
  prep_bn<<<1, 128, 0, stream>>>(dn3g, dn3b, dn3m, dn3v, scD3, shD3, 32);
  pad_bias<<<1, 64, 0, stream>>>(d_b4, b4p, 3, 16);
  f32_to_f16<<<cdiv((long)B * 32 * 32 * 3, 256), 256, 0, stream>>>(
      x, xh, (long)B * 32 * 32 * 3);

  // ---- encoder: stride-2 SAME 5x5 convs (pad_before = 1), relu+BN ----
  conv_gemm<2, true, 0><<<dim3((long)B * 256 / 64, 1), 128, 0, stream>>>(
      xh, wt1, a1, e_b1, sc1, sh1, 32, 32, 3, 16, 16, 5, 5, 2, 1, 1, 75, 96, 32);
  conv_gemm<4, false, 0><<<dim3((long)B * 64 / 64, 1), 128, 0, stream>>>(
      a1, wt2, a2, e_b2, sc2, sh2, 16, 16, 32, 8, 8, 5, 5, 2, 1, 1, 800, 800, 64);
  conv_gemm<8, false, 0><<<dim3((long)B * 16 / 64, 1), 128, 0, stream>>>(
      a2, wt3, a3, e_b3, sc3, sh3, 8, 8, 64, 4, 4, 5, 5, 2, 1, 1, 1600, 1600, 128);
  // ---- FC heads (as 1x1 conv with Cin=2048), relu ----
  conv_gemm<8, false, 1><<<dim3(B / 64, 1), 128, 0, stream>>>(
      a3, wtm, zm, e_bm, nullptr, nullptr, 1, 1, 2048, 1, 1, 1, 1, 1, 0, 1,
      2048, 2048, 128);
  conv_gemm<8, false, 1><<<dim3(B / 64, 1), 128, 0, stream>>>(
      a3, wtv, zlv, e_bv, nullptr, nullptr, 1, 1, 2048, 1, 1, 1, 1, 1, 0, 1,
      2048, 2048, 128);
  zsample<<<cdiv((long)B * 128, 256), 256, 0, stream>>>(zm, zlv, eps_z, z,
                                                        (long)B * 128);
  // ---- decoder dense 128 -> 8192, relu ----
  conv_gemm<8, false, 1><<<dim3(B / 64, 64), 128, 0, stream>>>(
      z, wtd, d0, d_bd, nullptr, nullptr, 1, 1, 128, 1, 1, 1, 1, 1, 0, 1, 128,
      128, 8192);
  // ---- dec1: 1x1 conv on [B,8,8,128], relu+BN ----
  conv_gemm<8, false, 0><<<dim3((long)B * 64 / 64, 1), 128, 0, stream>>>(
      d0, wtD1, d1, d_b1, scD1, shD1, 8, 8, 128, 8, 8, 1, 1, 1, 0, 1, 128, 128,
      128);
  // ---- dec2/dec3: 3x3 convT stride 2 as dil=2 gather conv, relu+BN ----
  conv_gemm<4, false, 0><<<dim3((long)B * 256 / 64, 1), 128, 0, stream>>>(
      d1, wtD2, d2, d_b2, scD2, shD2, 8, 8, 128, 16, 16, 3, 3, 1, 1, 2, 1152,
      1152, 64);
  conv_gemm<2, false, 0><<<dim3((long)B * 1024 / 64, 1), 128, 0, stream>>>(
      d2, wtD3, d3, d_b3, scD3, shD3, 16, 16, 64, 32, 32, 3, 3, 1, 1, 2, 576,
      576, 32);
  // ---- dec4: 3x3 SAME conv, sigmoid, Cout padded 3->16 ----
  conv_gemm<1, false, 2><<<dim3((long)B * 1024 / 64, 1), 128, 0, stream>>>(
      d3, wtD4, recon, b4p, nullptr, nullptr, 32, 32, 32, 32, 32, 3, 3, 1, 1,
      1, 288, 288, 16);
  // ---- error + Otsu threshold + mask ----
  otsu_mask<<<B, 256, 0, stream>>>(x, recon, mask_out);
}
